// NCA_19653770346819
// MI455X (gfx1250) — matrix operand order repaired and checked by reading, not compiled, not run on hardware
//
#include <hip/hip_runtime.h>

typedef __attribute__((ext_vector_type(16))) _Float16 v16h;
typedef __attribute__((ext_vector_type(8)))  _Float16 v8h;
typedef __attribute__((ext_vector_type(2)))  _Float16 v2h;
typedef __attribute__((ext_vector_type(8)))  float    v8f;
typedef __attribute__((ext_vector_type(4)))  float    v4f;
typedef __attribute__((ext_vector_type(2)))  float    v2f;

#define Bc    16
#define Hc    256
#define Wc    256
#define Cc    48
#define HID   128
#define KP    160   // 3*C padded 144 -> 160 (multiple of 32)
#define TPX   64    // pixels per block (one row segment)
#define TCOLS 66    // TPX + halo

// ---------------- weight pre-conversion: fp32 -> f16, transposed [N][K] ----------------
__global__ __launch_bounds__(256) void nca_cvt_weights(const float* __restrict__ w1,
                                                       const float* __restrict__ w2,
                                                       _Float16* __restrict__ w1h,
                                                       _Float16* __restrict__ w2h) {
  int i = blockIdx.x * 256 + threadIdx.x;
  const int N1 = HID * KP;   // 128 x 160
  const int N2 = Cc * HID;   // 48 x 128
  if (i < N1) {
    int n = i / KP, k = i % KP;
    w1h[i] = (k < 3 * Cc) ? (_Float16)w1[k * HID + n] : (_Float16)0.f;
  } else if (i < N1 + N2) {
    int j = i - N1;
    int n = j / HID, k = j % HID;
    w2h[j] = (_Float16)w2[k * Cc + n];
  }
}

// ---------------- main NCA step ----------------
__global__ __launch_bounds__(128) void nca_step(const float* __restrict__ x,
                                                const float* __restrict__ alive,
                                                const _Float16* __restrict__ w1h,
                                                const float* __restrict__ b1,
                                                const _Float16* __restrict__ w2h,
                                                const float* __restrict__ b2,
                                                float* __restrict__ out) {
  __shared__ alignas(16) _Float16 p_lds[TPX * KP];        // 20480 B: perception f16; h reuses wave slice
  __shared__ alignas(16) float    xt[3 * TCOLS * Cc];     // 38016 B: fp32 halo tile (stays live for residual)

  const int blk = blockIdx.x;
  const int x0  = (blk & 3) * TPX;        // 4 tiles across W
  const int y   = (blk >> 2) & (Hc - 1);
  const int b   = blk >> 10;
  const int tid = threadIdx.x;

  // ---- Stage A: cooperative halo load (zero 'SAME' padding), float4 ----
  const int NF4 = 3 * TCOLS * (Cc / 4);   // 2376
  for (int i = tid; i < NF4; i += 128) {
    int f4  = i % (Cc / 4);
    int rc  = i / (Cc / 4);
    int row = rc / TCOLS;
    int col = rc % TCOLS;
    int gy  = y + row - 1;
    int gx  = x0 + col - 1;
    v4f v = {0.f, 0.f, 0.f, 0.f};
    if ((unsigned)gy < (unsigned)Hc && (unsigned)gx < (unsigned)Wc) {
      v = *(const v4f*)(x + (((size_t)b * Hc + gy) * Wc + gx) * Cc + f4 * 4);
    }
    *(v4f*)(xt + (row * TCOLS + col) * Cc + f4 * 4) = v;
  }
  __syncthreads();

  // ---- Stage B: perception p = [x, sobel_x, sobel_y] -> f16 LDS, channel pairs ----
  for (int i = tid; i < TPX * (Cc / 2); i += 128) {
    int px = i / (Cc / 2);
    int c  = (i % (Cc / 2)) * 2;
    const float* ctr = xt + (1 * TCOLS + px + 1) * Cc + c;
    v2f a_mm = *(const v2f*)(ctr - TCOLS * Cc - Cc);
    v2f a_m0 = *(const v2f*)(ctr - TCOLS * Cc);
    v2f a_mp = *(const v2f*)(ctr - TCOLS * Cc + Cc);
    v2f a_0m = *(const v2f*)(ctr - Cc);
    v2f a_00 = *(const v2f*)(ctr);
    v2f a_0p = *(const v2f*)(ctr + Cc);
    v2f a_pm = *(const v2f*)(ctr + TCOLS * Cc - Cc);
    v2f a_p0 = *(const v2f*)(ctr + TCOLS * Cc);
    v2f a_pp = *(const v2f*)(ctr + TCOLS * Cc + Cc);
    v2f gxv = -a_mm + a_mp - 2.f * a_0m + 2.f * a_0p - a_pm + a_pp;
    v2f gyv = -a_mm - 2.f * a_m0 - a_mp + a_pm + 2.f * a_p0 + a_pp;
    *(v2h*)(p_lds + px * KP + c)          = __builtin_convertvector(a_00, v2h);
    *(v2h*)(p_lds + px * KP + Cc + c)     = __builtin_convertvector(gxv, v2h);
    *(v2h*)(p_lds + px * KP + 2 * Cc + c) = __builtin_convertvector(gyv, v2h);
  }
  {
    const v8h zero8 = {};
    for (int i = tid; i < TPX * 2; i += 128) {            // zero K padding 144..159
      int px = i >> 1, chunk = i & 1;
      *(v8h*)(p_lds + px * KP + 3 * Cc + chunk * 8) = zero8;
    }
  }
  __syncthreads();

  const int wave  = tid >> 5;
  const int lane  = tid & 31;
  const int nlo   = lane & 15;
  const int hiH   = lane >> 4;      // 0 / 1
  const int koffA = hiH * 16;       // A-matrix K split: contiguous k = 16h + s (matches B)
  const int koffB = hiH * 16;       // B-matrix K split: contiguous k = 16h + s (matches A)
  const int px0   = wave * 16;

  // ---- GEMM1: p[16x160] x w1[160x128] via v_wmma_f32_16x16x32_f16 ----
  v8f acc[8] = {};
  #pragma unroll
  for (int ks = 0; ks < KP / 32; ++ks) {
    const int k0 = ks * 32;
    const _Float16* arow = p_lds + (px0 + nlo) * KP + k0 + koffA;
    v8h alo = *(const v8h*)(arow);
    v8h ahi = *(const v8h*)(arow + 8);
    v16h afrag = __builtin_shufflevector(alo, ahi, 0,1,2,3,4,5,6,7,8,9,10,11,12,13,14,15);
    #pragma unroll
    for (int nt = 0; nt < 8; ++nt) {
      v16h bfrag = *(const v16h*)(w1h + (nt * 16 + nlo) * KP + k0 + koffB);
      acc[nt] = __builtin_amdgcn_wmma_f32_16x16x32_f16(
          false, afrag, false, bfrag, (short)0, acc[nt], false, false);
    }
  }

  // ---- bias + ReLU -> h (f16) staged in this wave's own (now dead) p_lds slice ----
  _Float16* hbase = p_lds + px0 * KP;   // 16*128 f16 = 4096 B, fits in the 5120 B slice
  #pragma unroll
  for (int nt = 0; nt < 8; ++nt) {
    float bv = b1[nt * 16 + nlo];
    #pragma unroll
    for (int r = 0; r < 8; ++r) {
      float hv = acc[nt][r] + bv;
      hv = hv > 0.f ? hv : 0.f;
      hbase[(r + hiH * 8) * HID + nt * 16 + nlo] = (_Float16)hv;
    }
  }
  __syncthreads();

  // ---- GEMM2: h[16x128] x w2[128x48] ----
  v8f acc2[3] = {};
  #pragma unroll
  for (int ks = 0; ks < HID / 32; ++ks) {
    const int k0 = ks * 32;
    const _Float16* arow = hbase + nlo * HID + k0 + koffA;
    v8h alo = *(const v8h*)(arow);
    v8h ahi = *(const v8h*)(arow + 8);
    v16h afrag = __builtin_shufflevector(alo, ahi, 0,1,2,3,4,5,6,7,8,9,10,11,12,13,14,15);
    #pragma unroll
    for (int nt = 0; nt < 3; ++nt) {
      v16h bfrag = *(const v16h*)(w2h + (nt * 16 + nlo) * HID + k0 + koffB);
      acc2[nt] = __builtin_amdgcn_wmma_f32_16x16x32_f16(
          false, afrag, false, bfrag, (short)0, acc2[nt], false, false);
    }
  }

  // ---- epilogue: dx = (d + b2) * alive ; out = (x + dx) * alive ----
  // residual x comes from the still-live fp32 LDS tile (no second global pass over x)
  #pragma unroll
  for (int nt = 0; nt < 3; ++nt) {
    int c = nt * 16 + nlo;
    float b2v = b2[c];
    #pragma unroll
    for (int r = 0; r < 8; ++r) {
      int px = px0 + r + hiH * 8;
      size_t g = ((size_t)b * Hc + y) * Wc + (x0 + px);
      float am  = alive[g];
      float xv  = xt[(1 * TCOLS + px + 1) * Cc + c];
      float dxv = (acc2[nt][r] + b2v) * am;
      out[g * Cc + c] = (xv + dxv) * am;
    }
  }
}

extern "C" void kernel_launch(void* const* d_in, const int* in_sizes, int n_in,
                              void* d_out, int out_size, void* d_ws, size_t ws_size,
                              hipStream_t stream) {
  const float* x     = (const float*)d_in[0];
  const float* alive = (const float*)d_in[1];
  const float* w1    = (const float*)d_in[2];
  const float* b1    = (const float*)d_in[3];
  const float* w2    = (const float*)d_in[4];
  const float* b2    = (const float*)d_in[5];

  _Float16* w1h = (_Float16*)d_ws;                                        // 128*160 f16
  _Float16* w2h = (_Float16*)((char*)d_ws + HID * KP * sizeof(_Float16)); // 48*128 f16

  const int ncvt = HID * KP + Cc * HID;
  nca_cvt_weights<<<(ncvt + 255) / 256, 256, 0, stream>>>(w1, w2, w1h, w2h);

  const int nblocks = Bc * Hc * (Wc / TPX);   // 16384
  nca_step<<<nblocks, 128, 0, stream>>>(x, alive, w1h, b1, w2h, b2, (float*)d_out);
}
